// SelfAttention_29978871726282
// MI455X (gfx1250) — matrix operand order repaired and checked by reading, not compiled, not run on hardware
//
#include <hip/hip_runtime.h>

// MI455X / gfx1250, wave32. Matrix math via v_wmma_f32_16x16x32_f16; f16-side
// LDS staging via GLOBAL_LOAD_ASYNC_TO_LDS_B128 (ASYNCcnt) when available.
typedef __attribute__((ext_vector_type(16))) _Float16 v16h;
typedef __attribute__((ext_vector_type(8)))  _Float16 v8h;
typedef __attribute__((ext_vector_type(8)))  float    v8f;
typedef __attribute__((ext_vector_type(4)))  float    f32x4;
typedef int v4i __attribute__((vector_size(16)));   // matches builtin param type

#define WMMA(a, b, c) __builtin_amdgcn_wmma_f32_16x16x32_f16( \
    false, (a), false, (b), (short)0, (c), false, false)

#if defined(__has_builtin)
#if __has_builtin(__builtin_amdgcn_global_load_async_to_lds_b128) && \
    __has_builtin(__builtin_amdgcn_s_wait_asynccnt)
#define USE_ASYNC_LDS 1
#endif
#endif
#ifndef USE_ASYNC_LDS
#define USE_ASYNC_LDS 0
#endif

__device__ __forceinline__ v16h pack16(v8h lo, v8h hi) {
  return __builtin_shufflevector(lo, hi, 0,1,2,3,4,5,6,7,8,9,10,11,12,13,14,15);
}

// 16-byte global -> LDS copy. Async path writes LDS directly from the memory
// pipe (no VGPR round-trip), tracked by ASYNCcnt.
__device__ __forceinline__ void copy16B(_Float16* lds_dst, const _Float16* gsrc) {
#if USE_ASYNC_LDS
  __builtin_amdgcn_global_load_async_to_lds_b128(
      (__attribute__((address_space(1))) v4i*)gsrc,
      (__attribute__((address_space(3))) v4i*)lds_dst,
      0, 0);
#else
  *(v8h*)lds_dst = *(const v8h*)gsrc;
#endif
}

__device__ __forceinline__ void wait_async() {
#if USE_ASYNC_LDS
  __builtin_amdgcn_s_wait_asynccnt(0);
#endif
}

// ---------------------------------------------------------------------------
// Kernel 1: fused QKV projection.  C[4096][3072] = x @ [Wq;Wkv]^T  (f32 in,
// f16 out).  Q,K stored (b,h,t,e) row-major; V stored (b,h,e,t) transposed so
// the attention PV-GEMM B-fragment is two contiguous 8-half loads.
// Block: 256 thr = 8 waves (4 M x 2 N), tile 128x128, K-step 32.
// fp32 sources need cvt in VALU, so staging stays on the register path here.
// ---------------------------------------------------------------------------
__global__ __launch_bounds__(256) void qkv_kernel(
    const float* __restrict__ x, const float* __restrict__ Wq,
    const float* __restrict__ Wkv,
    _Float16* __restrict__ Q, _Float16* __restrict__ Kb, _Float16* __restrict__ Vb)
{
  __shared__ _Float16 As[128 * 40];   // 128 rows x 32 k, stride 40 (16B align)
  __shared__ _Float16 Bs[128 * 40];
  const int tid  = threadIdx.x;
  const int lane = tid & 31, wid = tid >> 5;
  const int wm = wid & 3, wn = wid >> 2;
  const int l16 = lane & 15, half = lane >> 4, base = half * 8;
  const int m0 = blockIdx.y * 128;
  const int n0 = blockIdx.x * 128;   // 0..23 -> [0,3072); never straddles 1024s
  const float* Wsrc = (n0 < 1024) ? (Wq + (size_t)n0 * 1024)
                                  : (Wkv + (size_t)(n0 - 1024) * 1024);
  v8f acc[2][4] = {};

  for (int k0 = 0; k0 < 1024; k0 += 32) {
    __syncthreads();
    {
      const int c4 = (tid & 7) * 4;
      const int rb = tid >> 3;
#pragma unroll
      for (int p = 0; p < 4; ++p) {
        int r = rb + p * 32;
        f32x4 va = *(const f32x4*)(x + (size_t)(m0 + r) * 1024 + k0 + c4);
        f32x4 vb = *(const f32x4*)(Wsrc + (size_t)r * 1024 + k0 + c4);
        _Float16* da = &As[r * 40 + c4];
        _Float16* db = &Bs[r * 40 + c4];
#pragma unroll
        for (int j = 0; j < 4; ++j) { da[j] = (_Float16)va[j]; db[j] = (_Float16)vb[j]; }
      }
    }
    __syncthreads();

    v16h afr[2], bfr[4];
#pragma unroll
    for (int mi = 0; mi < 2; ++mi) {
      int row = wm * 32 + mi * 16 + l16;
      afr[mi] = pack16(*(const v8h*)&As[row * 40 + base],
                       *(const v8h*)&As[row * 40 + 16 + base]);
    }
#pragma unroll
    for (int ni = 0; ni < 4; ++ni) {
      int row = wn * 64 + ni * 16 + l16;
      int kk  = half * 8;
      bfr[ni] = pack16(*(const v8h*)&Bs[row * 40 + kk],
                       *(const v8h*)&Bs[row * 40 + kk + 16]);
    }
#pragma unroll
    for (int mi = 0; mi < 2; ++mi)
#pragma unroll
      for (int ni = 0; ni < 4; ++ni)
        acc[mi][ni] = WMMA(afr[mi], bfr[ni], acc[mi][ni]);
  }

  // scatter epilogue (branch is block-uniform: 128 | 1024)
#pragma unroll
  for (int mi = 0; mi < 2; ++mi) {
#pragma unroll
    for (int ni = 0; ni < 4; ++ni) {
      int n = n0 + wn * 64 + ni * 16 + l16;
#pragma unroll
      for (int r = 0; r < 8; ++r) {
        int m = m0 + wm * 32 + mi * 16 + r + half * 8;
        int b = m >> 11, t = m & 2047;
        _Float16 v = (_Float16)acc[mi][ni][r];
        if (n < 1024) {
          int h = n >> 6, e = n & 63;
          Q[(((size_t)(b * 16 + h)) * 2048 + t) * 64 + e] = v;
        } else if (n < 2048) {
          int nn = n - 1024, h = nn >> 6, e = nn & 63;
          Kb[(((size_t)(b * 16 + h)) * 2048 + t) * 64 + e] = v;
        } else {
          int nn = n - 2048, h = nn >> 6, e = nn & 63;
          Vb[(((size_t)(b * 16 + h)) * 64 + e) * 2048 + t] = v;  // transposed
        }
      }
    }
  }
}

// ---------------------------------------------------------------------------
// Kernel 2: flash attention. Block = 8 waves, each wave owns a 16-query tile;
// block streams 32-key chunks of K/V through double-buffered LDS staged with
// async copies: chunk i+1 transfers while chunk i is consumed by WMMAs.
// One s_wait_asynccnt + barrier per iteration.
// ---------------------------------------------------------------------------
__global__ __launch_bounds__(256) void attn_kernel(
    const _Float16* __restrict__ Qg, const _Float16* __restrict__ Kg,
    const _Float16* __restrict__ Vg, _Float16* __restrict__ Og)
{
  __shared__ _Float16 Ks[2][32 * 72];     // 32 keys x 64 e, stride 72
  __shared__ _Float16 Vs[2][64 * 40];     // 64 e x 32 keys, stride 40
  __shared__ _Float16 Ps[8 * 16 * 32];    // per-wave P scratch (16q x 32k)
  const int tid  = threadIdx.x, lane = tid & 31, wid = tid >> 5;
  const int l16 = lane & 15, half = lane >> 4, base = half * 8;
  const int bh = blockIdx.z * 16 + blockIdx.y;
  const int t0 = blockIdx.x * 128 + wid * 16;
  const size_t koff = (size_t)bh * 2048 * 64;
  const size_t voff = (size_t)bh * 64 * 2048;

  const int key = tid >> 3, kcol = (tid & 7) * 8;   // K-chunk staging slots
  const int ev  = tid >> 2, vcol = (tid & 3) * 8;   // V-chunk staging slots

  auto issue_chunk = [&](int kc, int buf) {
    copy16B(&Ks[buf][key * 72 + kcol],
            Kg + koff + (size_t)(kc + key) * 64 + kcol);
    copy16B(&Vs[buf][ev * 40 + vcol],
            Vg + voff + (size_t)ev * 2048 + kc + vcol);
  };

  // Q A-fragments (kept in registers for the whole key sweep)
  v16h aq[2];
  {
    const _Float16* qrow = Qg + koff + (size_t)(t0 + l16) * 64;
#pragma unroll
    for (int ks = 0; ks < 2; ++ks)
      aq[ks] = pack16(*(const v8h*)(qrow + ks * 32 + base),
                      *(const v8h*)(qrow + ks * 32 + 16 + base));
  }

  v8f o[4] = {};
  float mrun[8], lrun[8];
#pragma unroll
  for (int r = 0; r < 8; ++r) { mrun[r] = -1e30f; lrun[r] = 0.f; }

  issue_chunk(0, 0);                       // prologue prefetch
  for (int ci = 0; ci < 64; ++ci) {
    wait_async();          // my async copies into buf (ci&1) have landed
    __syncthreads();       // everyone's copies landed; prior reads of other buf done
    const _Float16* Kc = Ks[ci & 1];
    const _Float16* Vc = Vs[ci & 1];
    if (ci + 1 < 64) issue_chunk((ci + 1) * 32, (ci + 1) & 1);  // overlap w/ compute

    // S = Q (16x64) @ K_chunk^T (64x32): 2 n-subtiles x 2 k-steps
    v8f s[2] = {};
#pragma unroll
    for (int jn = 0; jn < 2; ++jn) {
      int krow = jn * 16 + l16;
#pragma unroll
      for (int ks = 0; ks < 2; ++ks) {
        int e0 = ks * 32 + half * 8;
        v16h bk = pack16(*(const v8h*)&Kc[krow * 72 + e0],
                         *(const v8h*)&Kc[krow * 72 + e0 + 16]);
        s[jn] = WMMA(aq[ks], bk, s[jn]);
      }
    }

    // online softmax; accum row r = query (r + 8*half), keys across l16
    float fac[8];
#pragma unroll
    for (int r = 0; r < 8; ++r) {
      float s0 = s[0][r] * 0.125f;         // e^-0.5 scale
      float s1 = s[1][r] * 0.125f;
      float mx = fmaxf(s0, s1);
#pragma unroll
      for (int off = 8; off >= 1; off >>= 1)
        mx = fmaxf(mx, __shfl_xor(mx, off, 16));
      float mnew = fmaxf(mrun[r], mx);
      float f  = __expf(mrun[r] - mnew);
      float p0 = __expf(s0 - mnew);
      float p1 = __expf(s1 - mnew);
      s[0][r] = p0; s[1][r] = p1;
      float rs = p0 + p1;
#pragma unroll
      for (int off = 8; off >= 1; off >>= 1)
        rs += __shfl_xor(rs, off, 16);
      lrun[r] = lrun[r] * f + rs;
      mrun[r] = mnew;
      fac[r]  = f;
    }
#pragma unroll
    for (int ni = 0; ni < 4; ++ni)
#pragma unroll
      for (int r = 0; r < 8; ++r) o[ni][r] *= fac[r];

    // reshape P: C-layout -> A-layout via per-wave LDS scratch (same-wave DS
    // ops are processed in-order on CDNA5, so no barrier needed)
    _Float16* pw = &Ps[wid * 512];
#pragma unroll
    for (int r = 0; r < 8; ++r) {
      int q = r + half * 8;
      pw[q * 32 + l16]      = (_Float16)s[0][r];
      pw[q * 32 + 16 + l16] = (_Float16)s[1][r];
    }
    const _Float16* pr = &Ps[wid * 512 + l16 * 32];
    v16h ap = pack16(*(const v8h*)(pr + base), *(const v8h*)(pr + 16 + base));

    // O += P (16x32) @ V_chunk (32x64)
#pragma unroll
    for (int ni = 0; ni < 4; ++ni) {
      int e = ni * 16 + l16;
      int k0 = half * 8;
      v16h bv = pack16(*(const v8h*)&Vc[e * 40 + k0],
                       *(const v8h*)&Vc[e * 40 + k0 + 16]);
      o[ni] = WMMA(ap, bv, o[ni]);
    }
  }

  // normalize and emit attn-out as (b, t, h*e) f16 for the final GEMM
  const int b = bh >> 4, h = bh & 15;
#pragma unroll
  for (int r = 0; r < 8; ++r) {
    float inv = 1.0f / lrun[r];
    int t = t0 + r + half * 8;
    _Float16* orow = Og + ((size_t)(b * 2048 + t)) * 1024 + h * 64;
#pragma unroll
    for (int ni = 0; ni < 4; ++ni)
      orow[ni * 16 + l16] = (_Float16)(o[ni][r] * inv);
  }
}

// ---------------------------------------------------------------------------
// Kernel 3: out = attn_out (f16 4096x1024) @ W_out^T + b_out, fp32 output.
// A-tile (f16 source) staged via async copies; W_out needs cvt so stays VALU.
// ---------------------------------------------------------------------------
__global__ __launch_bounds__(256) void outproj_kernel(
    const _Float16* __restrict__ A, const float* __restrict__ Wo,
    const float* __restrict__ bias, float* __restrict__ out)
{
  __shared__ _Float16 As[128 * 40];
  __shared__ _Float16 Bs[128 * 40];
  const int tid  = threadIdx.x;
  const int lane = tid & 31, wid = tid >> 5;
  const int wm = wid & 3, wn = wid >> 2;
  const int l16 = lane & 15, half = lane >> 4, base = half * 8;
  const int m0 = blockIdx.y * 128;
  const int n0 = blockIdx.x * 128;
  v8f acc[2][4] = {};

  for (int k0 = 0; k0 < 1024; k0 += 32) {
    __syncthreads();
    {
      // A already f16: 2 passes x 16B, async straight into LDS
      int c = (tid & 3) * 8, rb = tid >> 2;
#pragma unroll
      for (int p = 0; p < 2; ++p) {
        int r = rb + p * 64;
        copy16B(&As[r * 40 + c], A + (size_t)(m0 + r) * 1024 + k0 + c);
      }
      // W_out fp32 -> f16: 4 passes x float4 (register path, needs cvt)
      int c4 = (tid & 7) * 4, rb2 = tid >> 3;
#pragma unroll
      for (int p = 0; p < 4; ++p) {
        int r = rb2 + p * 32;
        f32x4 v = *(const f32x4*)(Wo + (size_t)(n0 + r) * 1024 + k0 + c4);
        _Float16* d = &Bs[r * 40 + c4];
#pragma unroll
        for (int j = 0; j < 4; ++j) d[j] = (_Float16)v[j];
      }
    }
    wait_async();
    __syncthreads();

    v16h afr[2], bfr[4];
#pragma unroll
    for (int mi = 0; mi < 2; ++mi) {
      int row = wm * 32 + mi * 16 + l16;
      afr[mi] = pack16(*(const v8h*)&As[row * 40 + base],
                       *(const v8h*)&As[row * 40 + 16 + base]);
    }
#pragma unroll
    for (int ni = 0; ni < 4; ++ni) {
      int row = wn * 64 + ni * 16 + l16;
      int kk = half * 8;
      bfr[ni] = pack16(*(const v8h*)&Bs[row * 40 + kk],
                       *(const v8h*)&Bs[row * 40 + kk + 16]);
    }
#pragma unroll
    for (int mi = 0; mi < 2; ++mi)
#pragma unroll
      for (int ni = 0; ni < 4; ++ni)
        acc[mi][ni] = WMMA(afr[mi], bfr[ni], acc[mi][ni]);
  }

#pragma unroll
  for (int mi = 0; mi < 2; ++mi) {
#pragma unroll
    for (int ni = 0; ni < 4; ++ni) {
      int n = n0 + wn * 64 + ni * 16 + l16;
      float bb = bias[n];
#pragma unroll
      for (int r = 0; r < 8; ++r) {
        int m = m0 + wm * 32 + mi * 16 + r + half * 8;
        out[(size_t)m * 1024 + n] = acc[mi][ni][r] + bb;
      }
    }
  }
}

// ---------------------------------------------------------------------------
extern "C" void kernel_launch(void* const* d_in, const int* in_sizes, int n_in,
                              void* d_out, int out_size, void* d_ws, size_t ws_size,
                              hipStream_t stream) {
  const float* x   = (const float*)d_in[0];
  const float* Wq  = (const float*)d_in[1];
  const float* Wkv = (const float*)d_in[2];
  const float* Wo  = (const float*)d_in[3];
  const float* bo  = (const float*)d_in[4];
  float* out = (float*)d_out;

  // workspace (f16): Q | K | V | attn_out, 8 MB each = 32 MB total
  _Float16* ws = (_Float16*)d_ws;
  const size_t SEG = (size_t)4 * 1024 * 1024;   // 4M halves
  _Float16* Q = ws;
  _Float16* K = ws + SEG;
  _Float16* V = ws + 2 * SEG;
  _Float16* O = ws + 3 * SEG;

  qkv_kernel<<<dim3(24, 32), 256, 0, stream>>>(x, Wq, Wkv, Q, K, V);
  attn_kernel<<<dim3(16, 16, 2), 256, 0, stream>>>(Q, K, V, O);
  outproj_kernel<<<dim3(8, 32), 256, 0, stream>>>(O, Wo, bo, out);
}
